// SimpleLidarEncoder_73160472920346
// MI455X (gfx1250) — matrix operand order repaired and checked by reading, not compiled, not run on hardware
//
#include <hip/hip_runtime.h>
#include <hip/hip_bf16.h>

typedef __attribute__((ext_vector_type(16))) _Float16 v16h;
typedef __attribute__((ext_vector_type(8)))  _Float16 v8h;
typedef __attribute__((ext_vector_type(8)))  float    v8f;

#define SEGS   64
#define DOUT   512
#define TILE_M 128
#define HSTRIDE 72   // padded f16 row stride for LDS h-buffer

// ---------------------------------------------------------------------------
// Zero the accumulator region of the workspace (sums[64*512] + counts[64]).
// ---------------------------------------------------------------------------
__global__ void lidar_zero_kernel(float* acc, int n) {
    int i = blockIdx.x * blockDim.x + threadIdx.x;
    if (i < n) acc[i] = 0.0f;
}

// ---------------------------------------------------------------------------
// Repack W2 [64][512] f32 (row-major, K-major) into f16 B-fragment order:
//   w2h[ ((ntile*2 + kh)*32 + lane)*16 + j ]
// B layout (16-bit, 32x16): lanes 0-15 hold K=0-7 & 16-23,
// lanes 16-31 hold K=8-15 & 24-31; column n = ntile*16 + (lane&15).
// ---------------------------------------------------------------------------
__global__ void lidar_prep_w2_kernel(const float* __restrict__ W2,
                                     _Float16* __restrict__ w2h) {
    int idx = blockIdx.x * blockDim.x + threadIdx.x;   // 0 .. 32767
    int j     = idx & 15;
    int lane  = (idx >> 4) & 31;
    int kh    = (idx >> 9) & 1;
    int ntile = idx >> 10;
    int n = ntile * 16 + (lane & 15);
    int k = kh * 32 + ((j < 8) ? ((lane >> 4) * 8 + j) : (16 + (lane >> 4) * 8 + (j - 8)));
    w2h[idx] = (_Float16)W2[k * DOUT + n];
}

// ---------------------------------------------------------------------------
// Fused: h = relu(points@W1+b1) in LDS (f16) ->
//        f = relu(h@W2+b2) via v_wmma_f32_16x16x32_f16 ->
//        per-segment column sums via in-register reduction + atomics.
// Block = 256 threads (8 waves), 128 points. Wave w owns columns [w*64, w*64+64).
// ---------------------------------------------------------------------------
__global__ void __launch_bounds__(256)
lidar_main_kernel(const float* __restrict__ points,
                  const int*   __restrict__ segids,
                  const float* __restrict__ W1,
                  const float* __restrict__ b1,
                  const float* __restrict__ b2,
                  const _Float16* __restrict__ w2h,
                  float* __restrict__ sums,
                  float* __restrict__ counts) {
    __shared__ _Float16 hbuf[TILE_M * HSTRIDE];

    const int tid  = threadIdx.x;
    const int base = blockIdx.x * TILE_M;         // first point of this block

    // ---- Phase 1: layer 1 into LDS (A-fragment friendly f16 layout) ----
    {
        const int k  = tid & 63;                  // output feature 0..63
        const int rg = tid >> 6;                  // row group 0..3
        const float w0 = W1[k], w1 = W1[64 + k], w2c = W1[128 + k], w3 = W1[192 + k];
        const float bb = b1[k];
        #pragma unroll 4
        for (int i = 0; i < 32; ++i) {
            const int row = rg * 32 + i;
            const float4 p = ((const float4*)points)[base + row];
            float hv = fmaf(p.x, w0, fmaf(p.y, w1, fmaf(p.z, w2c, fmaf(p.w, w3, bb))));
            hv = fmaxf(hv, 0.0f);
            hbuf[row * HSTRIDE + k] = (_Float16)hv;
        }
    }

    // Segment-uniformity test for this block (segment_ids are sorted).
    const int seg_first = segids[base];
    const int seg_last  = segids[base + TILE_M - 1];
    const bool fast = (seg_first == seg_last);

    __syncthreads();

    // ---- Phase 2: WMMA layer 2 + fused pooling ----
    const int wave  = tid >> 5;
    const int lane  = tid & 31;
    const int lhalf = lane >> 4;                  // 0: lanes 0-15, 1: lanes 16-31
    const int lrow  = lane & 15;
    const int nbase = wave * 64;                  // this wave's column slice

    // Preload all B fragments for this wave's 4 N-tiles x 2 K-steps.
    v16h Bf[4][2];
    #pragma unroll
    for (int t = 0; t < 4; ++t)
        #pragma unroll
        for (int kh = 0; kh < 2; ++kh) {
            const int frag = ((wave * 4 + t) * 2 + kh);
            Bf[t][kh] = *(const v16h*)(w2h + (frag * 32 + lane) * 16);
        }

    // Per-lane bias for the column this lane owns in each N-tile.
    float b2c[4];
    #pragma unroll
    for (int t = 0; t < 4; ++t) b2c[t] = b2[nbase + t * 16 + lrow];

    float colsum[4] = {0.0f, 0.0f, 0.0f, 0.0f};

    #pragma unroll
    for (int mt = 0; mt < 8; ++mt) {
        // A fragments from LDS, per ISA 16-bit A 16x32 layout:
        // lane<16: K 0-7 & 16-23 ; lane>=16: K 8-15 & 24-31 (row = lrow).
        const int m = mt * 16 + lrow;
        v16h Af[2];
        #pragma unroll
        for (int kh = 0; kh < 2; ++kh) {
            const int off = m * HSTRIDE + kh * 32 + lhalf * 8;
            v8h lo = *(const v8h*)(hbuf + off);
            v8h hi = *(const v8h*)(hbuf + off + 16);
            Af[kh] = __builtin_shufflevector(lo, hi,
                        0, 1, 2, 3, 4, 5, 6, 7, 8, 9, 10, 11, 12, 13, 14, 15);
        }

        #pragma unroll
        for (int t = 0; t < 4; ++t) {
            v8f d = {};
            d = __builtin_amdgcn_wmma_f32_16x16x32_f16(false, Af[0], false, Bf[t][0],
                                                       (short)0, d, false, false);
            d = __builtin_amdgcn_wmma_f32_16x16x32_f16(false, Af[1], false, Bf[t][1],
                                                       (short)0, d, false, false);
            if (fast) {
                #pragma unroll
                for (int v = 0; v < 8; ++v)
                    colsum[t] += fmaxf(d[v] + b2c[t], 0.0f);
            } else {
                // Rare segment-boundary block: per-element atomics.
                const int col = nbase + t * 16 + lrow;
                #pragma unroll
                for (int v = 0; v < 8; ++v) {
                    const int mloc = (lane < 16) ? v : (v + 8);
                    const int row  = base + mt * 16 + mloc;
                    const int seg  = segids[row];
                    atomicAdd(&sums[seg * DOUT + col], fmaxf(d[v] + b2c[t], 0.0f));
                }
            }
        }
    }

    if (fast) {
        // Combine the two row-halves (lane n <-> lane n+16 hold rows 0-7 / 8-15
        // of the same column), then one atomic per column.
        #pragma unroll
        for (int t = 0; t < 4; ++t)
            colsum[t] += __shfl_xor(colsum[t], 16, 32);
        if (lane < 16) {
            #pragma unroll
            for (int t = 0; t < 4; ++t)
                atomicAdd(&sums[seg_first * DOUT + nbase + t * 16 + lrow], colsum[t]);
        }
    }

    if (tid == 0) {
        if (fast) {
            atomicAdd(&counts[seg_first], (float)TILE_M);
        } else {
            for (int r = 0; r < TILE_M; ++r)
                atomicAdd(&counts[segids[base + r]], 1.0f);
        }
    }
}

// ---------------------------------------------------------------------------
// mean = sums / max(counts, 1)
// ---------------------------------------------------------------------------
__global__ void lidar_finalize_kernel(const float* __restrict__ sums,
                                      const float* __restrict__ counts,
                                      float* __restrict__ out) {
    int i = blockIdx.x * blockDim.x + threadIdx.x;   // 0 .. 32767
    float c = fmaxf(counts[i >> 9], 1.0f);
    out[i] = sums[i] / c;
}

extern "C" void kernel_launch(void* const* d_in, const int* in_sizes, int n_in,
                              void* d_out, int out_size, void* d_ws, size_t ws_size,
                              hipStream_t stream) {
    const float* points = (const float*)d_in[0];
    const int*   segids = (const int*)d_in[1];
    const float* W1     = (const float*)d_in[2];
    const float* b1     = (const float*)d_in[3];
    const float* W2     = (const float*)d_in[4];
    const float* b2     = (const float*)d_in[5];

    const int N = in_sizes[0] / 4;                 // number of points (524288)
    const int nblocks = N / TILE_M;                // 4096

    // Workspace layout: sums[64*512] f32 | counts[64] f32 | (align 256) | w2h f16
    float*    sums   = (float*)d_ws;
    float*    counts = sums + SEGS * DOUT;
    _Float16* w2h    = (_Float16*)((char*)d_ws + 132096);  // 256B-aligned

    const int accN = SEGS * DOUT + SEGS;
    lidar_zero_kernel<<<(accN + 255) / 256, 256, 0, stream>>>(sums, accN);
    lidar_prep_w2_kernel<<<(SEGS * DOUT) / 256, 256, 0, stream>>>(W2, w2h);
    lidar_main_kernel<<<nblocks, 256, 0, stream>>>(points, segids, W1, b1, b2,
                                                   w2h, sums, counts);
    lidar_finalize_kernel<<<(SEGS * DOUT) / 256, 256, 0, stream>>>(sums, counts,
                                                                  (float*)d_out);
}
